// SpectraSQUADTokenModel_41077067219028
// MI455X (gfx1250) — matrix operand order repaired and checked by reading, not compiled, not run on hardware
//
#include <hip/hip_runtime.h>
#include <hip/hip_bf16.h>

// ---------------------------------------------------------------------------
// SpectraSQUADTokenModel for MI455X (gfx1250, wave32)
//
// Roofline: context_reps is 64*4096*768*4 B = 805 MB -> 34.6 us at 23.3 TB/s.
// Since LQ==1, score = x.(Wc@q) + q.bc, so the 51.5 GFLOP GEMM folds into a
// 0.4 GFLOP batched GEMV -> purely memory bound. The GEMV is executed on the
// WMMA pipe (v_wmma_f32_16x16x32_bf16) with u packed into column 0 of B;
// the wasted N-lanes are free against the HBM floor.
// ---------------------------------------------------------------------------

#define B_      64
#define LC_     4096
#define H_      768
#define P_      128
#define MAXLEN_ 4096
#define BUDGET_ 200.0f
#define MAX_ITER_ 100

typedef __attribute__((ext_vector_type(16))) __bf16 v16bf;
typedef __attribute__((ext_vector_type(8)))  float  v8f;

// ---------------------------------------------------------------------------
// Kernel A: per batch b:
//   q[p]   = sum_h qreps[b,h]*Wq[h,p] + bq[p]        (P=128)
//   u[h]   = sum_p Wc[h,p]*q[p]                      (H=768)
//   qbc    = sum_p q[p]*bc[p]
// ---------------------------------------------------------------------------
__global__ __launch_bounds__(256) void proj_kernel(
    const float* __restrict__ qreps, const float* __restrict__ Wq,
    const float* __restrict__ bq,    const float* __restrict__ Wc,
    const float* __restrict__ bc,    float* __restrict__ U,
    float* __restrict__ qbc)
{
  __shared__ float q_s[P_];
  const int b = blockIdx.x, t = threadIdx.x;
  const float* qr = qreps + (size_t)b * H_;
  if (t < P_) {
    float acc = bq[t];
    for (int h = 0; h < H_; ++h)
      acc = fmaf(qr[h], Wq[(size_t)h * P_ + t], acc);  // coalesced in t
    q_s[t] = acc;
  }
  __syncthreads();
  for (int h = t; h < H_; h += 256) {
    const float* wr = Wc + (size_t)h * P_;
    float acc = 0.f;
    #pragma unroll 8
    for (int p = 0; p < P_; ++p) acc = fmaf(wr[p], q_s[p], acc);
    U[(size_t)b * H_ + h] = acc;
  }
  if (t == 0) {
    float acc = 0.f;
    for (int p = 0; p < P_; ++p) acc = fmaf(q_s[p], bc[p], acc);
    qbc[b] = acc;
  }
}

// ---------------------------------------------------------------------------
// Kernel B: Sraw[b,j] = context_reps[b,j,:] . U[b,:]  via WMMA.
// One wave per 16-row j-tile. A = bf16(x tile 16x32), B column 0 = bf16(u),
// D column 0 (lane 0 -> M0..7, lane 16 -> M8..15) = the 16 scores.
// A layout (16-bit 16x32): lanes 0-15 hold K 0..7 & 16..23; lanes 16-31 hold
// K 8..15 & 24..31. B layout (32x16): lane n<16 = col n K0..7,16..23; lane n+16 =
// col n K8..15,24..31.
// ---------------------------------------------------------------------------
__global__ __launch_bounds__(256) void score_wmma_kernel(
    const float* __restrict__ X,   // [B, LC, H]
    const float* __restrict__ U,   // [B, H]
    float* __restrict__ Sraw)      // [B, LC]
{
  const int b    = blockIdx.y;
  const int wave = threadIdx.x >> 5;
  const int lane = threadIdx.x & 31;
  const int half = lane >> 4;        // 0: lanes 0-15, 1: lanes 16-31
  const int m    = lane & 15;
  const int j0   = (blockIdx.x * 8 + wave) * 16;

  const float* xrow = X + ((size_t)b * LC_ + (size_t)(j0 + m)) * H_;
  const float* ub   = U + (size_t)b * H_;
  const float  sel  = (m == 0) ? 1.f : 0.f;   // B-matrix has only column 0

  v8f acc = {};
  for (int k = 0; k < H_; k += 32) {
    // next-tile prefetch of this row's stream (global_prefetch_b8)
    if (k + 96 <= H_) __builtin_prefetch(xrow + k + 64, 0, 1);

    const float4* pa0 = (const float4*)(xrow + k + half * 8);       // K lo chunk
    const float4* pa1 = (const float4*)(xrow + k + 16 + half * 8);  // K hi chunk
    float4 f0 = pa0[0], f1 = pa0[1], f2 = pa1[0], f3 = pa1[1];

    v16bf a;
    a[0]=(__bf16)f0.x;  a[1]=(__bf16)f0.y;  a[2]=(__bf16)f0.z;  a[3]=(__bf16)f0.w;
    a[4]=(__bf16)f1.x;  a[5]=(__bf16)f1.y;  a[6]=(__bf16)f1.z;  a[7]=(__bf16)f1.w;
    a[8]=(__bf16)f2.x;  a[9]=(__bf16)f2.y;  a[10]=(__bf16)f2.z; a[11]=(__bf16)f2.w;
    a[12]=(__bf16)f3.x; a[13]=(__bf16)f3.y; a[14]=(__bf16)f3.z; a[15]=(__bf16)f3.w;

    const float* up = ub + k + half * 8;  // lane0: K0..7,16..23, lane16: K8..15,24..31
    float4 u0 = *(const float4*)(up);
    float4 u1 = *(const float4*)(up + 4);
    float4 u2 = *(const float4*)(up + 16);
    float4 u3 = *(const float4*)(up + 20);

    v16bf bv;
    bv[0]=(__bf16)(u0.x*sel);  bv[1]=(__bf16)(u0.y*sel);
    bv[2]=(__bf16)(u0.z*sel);  bv[3]=(__bf16)(u0.w*sel);
    bv[4]=(__bf16)(u1.x*sel);  bv[5]=(__bf16)(u1.y*sel);
    bv[6]=(__bf16)(u1.z*sel);  bv[7]=(__bf16)(u1.w*sel);
    bv[8]=(__bf16)(u2.x*sel);  bv[9]=(__bf16)(u2.y*sel);
    bv[10]=(__bf16)(u2.z*sel); bv[11]=(__bf16)(u2.w*sel);
    bv[12]=(__bf16)(u3.x*sel); bv[13]=(__bf16)(u3.y*sel);
    bv[14]=(__bf16)(u3.z*sel); bv[15]=(__bf16)(u3.w*sel);

    acc = __builtin_amdgcn_wmma_f32_16x16x32_bf16(
        /*neg_a=*/false, a, /*neg_b=*/false, bv,
        /*c_mod=*/(short)0, acc, /*reuse_a=*/false, /*reuse_b=*/false);
  }

  if (m == 0) {  // lane 0 holds M 0..7 of column 0, lane 16 holds M 8..15
    float* o = Sraw + (size_t)b * LC_ + j0 + half * 8;
    #pragma unroll
    for (int r = 0; r < 8; ++r) o[r] = acc[r];
  }
}

// ---------------------------------------------------------------------------
// Kernel C: mask/penalty, exact bisection budget projection, token_z gather.
// One 256-thread (8-wave) block per batch; s lives in LDS (16 KB).
// ---------------------------------------------------------------------------
__device__ __forceinline__ float wave_sum(float v) {
  #pragma unroll
  for (int o = 16; o > 0; o >>= 1) v += __shfl_down(v, o, 32);
  return v;
}
__device__ __forceinline__ float wave_min(float v) {
  #pragma unroll
  for (int o = 16; o > 0; o >>= 1) v = fminf(v, __shfl_down(v, o, 32));
  return v;
}
__device__ __forceinline__ float wave_max(float v) {
  #pragma unroll
  for (int o = 16; o > 0; o >>= 1) v = fmaxf(v, __shfl_down(v, o, 32));
  return v;
}

__device__ __forceinline__ float block_sum(float v, float* red) {
  const int lane = threadIdx.x & 31, wid = threadIdx.x >> 5;
  v = wave_sum(v);
  if (lane == 0) red[wid] = v;
  __syncthreads();
  float r = 0.f;
  #pragma unroll
  for (int w = 0; w < 8; ++w) r += red[w];
  __syncthreads();
  return r;
}
__device__ __forceinline__ float block_min(float v, float* red) {
  const int lane = threadIdx.x & 31, wid = threadIdx.x >> 5;
  v = wave_min(v);
  if (lane == 0) red[wid] = v;
  __syncthreads();
  float r = red[0];
  #pragma unroll
  for (int w = 1; w < 8; ++w) r = fminf(r, red[w]);
  __syncthreads();
  return r;
}
__device__ __forceinline__ float block_max(float v, float* red) {
  const int lane = threadIdx.x & 31, wid = threadIdx.x >> 5;
  v = wave_max(v);
  if (lane == 0) red[wid] = v;
  __syncthreads();
  float r = red[0];
  #pragma unroll
  for (int w = 1; w < 8; ++w) r = fmaxf(r, red[w]);
  __syncthreads();
  return r;
}

__global__ __launch_bounds__(256) void project_kernel(
    const float* __restrict__ Sraw, const float* __restrict__ qbc,
    const int* __restrict__ mask,   const int* __restrict__ qe_arr,
    float* __restrict__ out)
{
  __shared__ float s_s[LC_];
  __shared__ float red[8];
  const int b = blockIdx.x, t = threadIdx.x;
  const float qb = qbc[b];

  float cnt = 0.f, vmin = 3.4e38f, vmax = -3.4e38f, sum0 = 0.f;
  for (int j = t; j < LC_; j += 256) {
    const int   mk = mask[(size_t)b * LC_ + j];
    const float s  = Sraw[(size_t)b * LC_ + j] + qb + ((float)mk - 1.f) * 100000.f;
    s_s[j] = s;
    cnt  += (float)mk;
    vmin  = fminf(vmin, s);
    vmax  = fmaxf(vmax, s);
    sum0 += fminf(fmaxf(s, 0.f), 1.f);
  }
  const float ctx = block_sum(cnt, red) - 1.f;   // context_length (exact in f32)
  vmin = block_min(vmin, red);
  vmax = block_max(vmax, red);
  sum0 = block_sum(sum0, red);

  const bool feasible = (sum0 <= BUDGET_);
  float lo = vmin - 1.f, hi = vmax;
  for (int it = 0; it < MAX_ITER_; ++it) {
    const float mid = 0.5f * (lo + hi);
    float zs = 0.f;
    #pragma unroll 4
    for (int j = t; j < LC_; j += 256)
      zs += fminf(fmaxf(s_s[j] - mid, 0.f), 1.f);
    zs = block_sum(zs, red);          // block-uniform -> no divergence below
    if (zs > BUDGET_) lo = mid; else hi = mid;
  }
  const float tau = feasible ? 0.f : 0.5f * (lo + hi);

  for (int j = t; j < LC_; j += 256)
    s_s[j] = fminf(fmaxf(s_s[j] - tau, 0.f), 1.f);   // z_probs in place
  __syncthreads();

  const int qe = qe_arr[b];
  const int ctxi = (int)ctx;
  for (int p = t; p < MAXLEN_; p += 256) {
    const int idx  = p - qe + 1;
    const int idxc = idx < 0 ? 0 : (idx > LC_ - 1 ? LC_ - 1 : idx);
    const float g  = s_s[idxc];
    const bool in_ctx = (idx >= 1) && (idx <= ctxi);
    out[(size_t)b * MAXLEN_ + p] = (p < qe) ? 1.f : (in_ctx ? g : 0.f);
  }
}

// ---------------------------------------------------------------------------
extern "C" void kernel_launch(void* const* d_in, const int* in_sizes, int n_in,
                              void* d_out, int out_size, void* d_ws, size_t ws_size,
                              hipStream_t stream) {
  const float* qreps = (const float*)d_in[0];  // [B,1,H]
  const float* creps = (const float*)d_in[1];  // [B,LC,H]
  const float* Wq    = (const float*)d_in[2];  // [H,P]
  const float* bq    = (const float*)d_in[3];  // [P]
  const float* Wc    = (const float*)d_in[4];  // [H,P]
  const float* bc    = (const float*)d_in[5];  // [P]
  const int*   mask  = (const int*)d_in[6];    // [B,LC]
  const int*   qe    = (const int*)d_in[7];    // [B]
  float* out = (float*)d_out;                  // [B,MAXLEN]

  float* U    = (float*)d_ws;                  // [B,H]   = 49152 floats
  float* qbc  = U + (size_t)B_ * H_;           // [B]     (padded to 256)
  float* Sraw = qbc + 256;                     // [B,LC]  = 262144 floats

  proj_kernel<<<B_, 256, 0, stream>>>(qreps, Wq, bq, Wc, bc, U, qbc);
  score_wmma_kernel<<<dim3(LC_ / (16 * 8), B_), 256, 0, stream>>>(creps, U, Sraw);
  project_kernel<<<B_, 256, 0, stream>>>(Sraw, qbc, mask, qe, out);
}
